// EDDA_21852793602799
// MI455X (gfx1250) — matrix operand order repaired and checked by reading, not compiled, not run on hardware
//
#include <hip/hip_runtime.h>
#include <hip/hip_bf16.h>

typedef __attribute__((ext_vector_type(16))) __bf16 v16bf;
typedef __attribute__((ext_vector_type(8)))  __bf16 v8bf;
typedef __attribute__((ext_vector_type(8)))  float  v8f;
typedef __attribute__((ext_vector_type(4))) unsigned int u32x4;
typedef __attribute__((ext_vector_type(8))) int          i32x8;
typedef __attribute__((ext_vector_type(4))) int          i32x4;

#define EPSF 1e-5f

#if defined(__gfx1250__) && __has_builtin(__builtin_amdgcn_tensor_load_to_lds) && \
    __has_builtin(__builtin_amdgcn_s_wait_tensorcnt)
#define HAS_TDM 1
#else
#define HAS_TDM 0
#endif

__device__ __forceinline__ float gelu_f(float v) {
    return 0.5f * v * (1.0f + erff(v * 0.70710678118654752f));
}
__device__ __forceinline__ float sigmoid_f(float v) {
    return 1.0f / (1.0f + expf(-v));
}

// ---------------------------------------------------------------------------
// K0: one-time fp32 -> bf16 conversion of the pointwise weight (256x256).
// ---------------------------------------------------------------------------
__global__ void k0_cvt(const float* __restrict__ pw, __bf16* __restrict__ pwb) {
    int i = blockIdx.x * 256 + threadIdx.x;
    pwb[i] = (__bf16)pw[i];
}

// ---------------------------------------------------------------------------
// K1: one pass over x -> row means (y_pool), col means (x_pool), gap.
// One block per (b,c) plane (64x64). LDS tile is swizzled tile[h][ (w+h)&63 ]
// so both the row-sum pass and the column-sum pass are bank-conflict free.
// ---------------------------------------------------------------------------
__global__ void k1_reduce(const float* __restrict__ x,
                          float* __restrict__ xpool, float* __restrict__ ypool,
                          float* __restrict__ gap) {
    const int bc = blockIdx.x;      // b*256 + c
    const int t  = threadIdx.x;     // 256 threads
    __shared__ float tile[4096];
    __shared__ float rs[64];
    const float* p = x + (size_t)bc * 4096;
    const int h  = t >> 2;
    const int w0 = (t & 3) * 16;
#pragma unroll
    for (int j = 0; j < 16; j += 4) {
        float4 v4 = *(const float4*)(p + h * 64 + w0 + j);
        tile[h * 64 + ((w0 + j + 0 + h) & 63)] = v4.x;
        tile[h * 64 + ((w0 + j + 1 + h) & 63)] = v4.y;
        tile[h * 64 + ((w0 + j + 2 + h) & 63)] = v4.z;
        tile[h * 64 + ((w0 + j + 3 + h) & 63)] = v4.w;
    }
    __syncthreads();
    if (t < 64) {
        float rsum = 0.f, csum = 0.f;
#pragma unroll 8
        for (int i = 0; i < 64; ++i) {
            rsum += tile[t * 64 + ((i + t) & 63)];   // row t  (mean over w)
            csum += tile[i * 64 + ((t + i) & 63)];   // col t  (mean over h)
        }
        ypool[(size_t)bc * 64 + t] = rsum * (1.f / 64.f);
        xpool[(size_t)bc * 64 + t] = csum * (1.f / 64.f);
        rs[t] = rsum;
    }
    __syncthreads();
    if (t == 0) {
        float s = 0.f;
#pragma unroll
        for (int i = 0; i < 64; ++i) s += rs[i];
        gap[bc] = s * (1.f / 4096.f);
    }
}

// ---------------------------------------------------------------------------
// K2: per-batch tiny work: axis_conv (both axes), kernel-mix softmax kw,
// dilated 1-D conv lines, grouped channel attention. One block per batch.
// ---------------------------------------------------------------------------
__global__ void k2_small(const float* __restrict__ xpool, const float* __restrict__ ypool,
                         const float* __restrict__ gap,
                         const float* adww, const float* adwb,
                         const float* c4w, const float* c4b,
                         const float* w41, const float* b41,
                         const float* kgw1, const float* kgb1,
                         const float* kgw2, const float* kgb2,
                         const float* dxw, const float* dyw,
                         const float* gcw1, const float* gcb1,
                         const float* gcw2, const float* gcb2,
                         float* __restrict__ xline, float* __restrict__ yline,
                         float* __restrict__ cha) {
    const int b = blockIdx.x, t = threadIdx.x;   // 256 threads
    __shared__ float gl[256], xl[64], yl[64], hid[64], kgh[16], kwv[3];
    gl[t] = gap[b * 256 + t];
    __syncthreads();

    if (t < 128) {
        // axis_conv: per-channel affine -> 1x1 (C->4) -> GELU -> 1x1 (4->1)
        const int l = t & 63;
        const float* pool = ((t < 64) ? xpool : ypool) + (size_t)b * 16384 + l;
        float a0 = c4b[0], a1 = c4b[1], a2 = c4b[2], a3 = c4b[3];
        for (int c = 0; c < 256; ++c) {
            float tv = pool[c * 64] * adww[c] + adwb[c];
            a0 += c4w[c]         * tv;
            a1 += c4w[256 + c]   * tv;
            a2 += c4w[512 + c]   * tv;
            a3 += c4w[768 + c]   * tv;
        }
        float line = b41[0] + w41[0] * gelu_f(a0) + w41[1] * gelu_f(a1)
                   + w41[2] * gelu_f(a2) + w41[3] * gelu_f(a3);
        if (t < 64) xl[l] = line; else yl[l] = line;
    } else if (t < 144) {
        const int m = t - 128;
        float a = kgb1[m];
        for (int c = 0; c < 256; ++c) a += gl[c] * kgw1[m * 256 + c];
        kgh[m] = gelu_f(a);
    } else if (t >= 192) {
        const int g = t - 192;
        float a = gcb1[g];
        for (int c = 0; c < 256; ++c) a += gl[c] * gcw1[g * 256 + c];
        hid[g] = gelu_f(a);
    }
    __syncthreads();

    if (t == 0) {  // softmax over 3 logits
        float lg[3], mx = -1e30f;
        for (int i = 0; i < 3; ++i) {
            float a = kgb2[i];
            for (int m = 0; m < 16; ++m) a += kgh[m] * kgw2[i * 16 + m];
            lg[i] = a; mx = fmaxf(mx, a);
        }
        float s = 0.f;
        for (int i = 0; i < 3; ++i) { lg[i] = expf(lg[i] - mx); s += lg[i]; }
        for (int i = 0; i < 3; ++i) kwv[i] = lg[i] / s;
    }
    {   // grouped channel attention, c = t
        float a = gcb2[t];
        for (int g = 0; g < 64; ++g) a += hid[g] * gcw2[t * 64 + g];
        cha[b * 256 + t] = sigmoid_f(a);
    }
    __syncthreads();

    if (t < 128) {  // softmax-weighted dilated conv1d banks
        const int l = t & 63;
        const float* sl  = (t < 64) ? xl  : yl;
        const float* dwk = (t < 64) ? dxw : dyw;
        float acc = 0.f;
        for (int i = 0; i < 3; ++i) {
            int d = i + 1; float s = 0.f;
            for (int j = 0; j < 3; ++j) {
                int idx = l + (j - 1) * d;
                if ((unsigned)idx < 64u) s += dwk[i * 3 + j] * sl[idx];
            }
            acc += kwv[i] * s;
        }
        if (t < 64) xline[b * 64 + l] = acc; else yline[b * 64 + l] = acc;
    }
}

// ---------------------------------------------------------------------------
// K3: spatial attention. One block per batch. Mid tensor t (4,64,64) kept in
// LDS as bf16 (32 KB) between the two 5x5 convs.
// ---------------------------------------------------------------------------
__global__ void k3_spatial(const float* __restrict__ xline, const float* __restrict__ yline,
                           const float* prw1, const float* prb1,
                           const float* prw2, const float* prb2,
                           float* __restrict__ satt) {
    const int b = blockIdx.x, t = threadIdx.x;   // 256 threads
    __shared__ __bf16 tm[4][64][64];
    __shared__ float xls[64], yls[64];
    if (t < 64) xls[t] = xline[b * 64 + t];
    else if (t < 128) yls[t - 64] = yline[b * 64 + (t - 64)];
    __syncthreads();
    for (int r = 0; r < 16; ++r) {
        int pos = r * 256 + t, h = pos >> 6, w = pos & 63;
#pragma unroll
        for (int m = 0; m < 4; ++m) {
            float acc = prb1[m];
            for (int kh = 0; kh < 5; ++kh) {
                int hh = h + kh - 2;
                if ((unsigned)hh >= 64u) continue;
                float yv = yls[hh];
                for (int kw = 0; kw < 5; ++kw) {
                    int wi = w + kw - 2;
                    if ((unsigned)wi >= 64u) continue;
                    acc += prw1[((m * 2 + 0) * 5 + kh) * 5 + kw] * xls[wi]
                         + prw1[((m * 2 + 1) * 5 + kh) * 5 + kw] * yv;
                }
            }
            tm[m][h][w] = (__bf16)gelu_f(acc);
        }
    }
    __syncthreads();
    for (int r = 0; r < 16; ++r) {
        int pos = r * 256 + t, h = pos >> 6, w = pos & 63;
        float acc = prb2[0];
        for (int kh = 0; kh < 5; ++kh) {
            int hh = h + kh - 2;
            if ((unsigned)hh >= 64u) continue;
            for (int kw = 0; kw < 5; ++kw) {
                int wi = w + kw - 2;
                if ((unsigned)wi >= 64u) continue;
#pragma unroll
                for (int m = 0; m < 4; ++m)
                    acc += prw2[(m * 5 + kh) * 5 + kw] * (float)tm[m][hh][wi];
            }
        }
        satt[(size_t)b * 4096 + pos] = sigmoid_f(acc);
    }
}

// ---------------------------------------------------------------------------
// K4: fused  x*(1+spatial)*(1+channel) -> depthwise 3x3 -> BN1 -> GELU,
// emitting bf16 activations for the WMMA GEMM. One block per (b,c) plane.
// ---------------------------------------------------------------------------
__global__ void k4_fuse_dw(const float* __restrict__ x, const float* __restrict__ satt,
                           const float* __restrict__ cha, const float* __restrict__ dww,
                           const float* bn1g, const float* bn1b,
                           const float* bn1m, const float* bn1v,
                           __bf16* __restrict__ act) {
    const int bc = blockIdx.x, t = threadIdx.x;   // 256 threads
    const int b = bc >> 8, c = bc & 255;
    __shared__ float tile[64 * 65];               // pitch 65: conflict-free
    const float ca   = 1.f + cha[b * 256 + c];
    const float iv1  = bn1g[c] * rsqrtf(bn1v[c] + EPSF);
    const float bt1  = bn1b[c] - bn1m[c] * iv1;
    const float* xp = x + (size_t)bc * 4096;
    const float* sa = satt + (size_t)b * 4096;
    const int h = t >> 2, w0 = (t & 3) * 16;
#pragma unroll
    for (int j = 0; j < 16; ++j) {
        int idx = h * 64 + w0 + j;
        tile[h * 65 + w0 + j] = xp[idx] * (1.f + sa[idx]) * ca;
    }
    __syncthreads();
    float wk[9];
#pragma unroll
    for (int i = 0; i < 9; ++i) wk[i] = dww[c * 9 + i];
    __bf16* ap = act + (size_t)bc * 4096 + h * 64 + w0;
#pragma unroll
    for (int j = 0; j < 16; ++j) {
        int w = w0 + j;
        float acc = 0.f;
#pragma unroll
        for (int kh = 0; kh < 3; ++kh) {
            int hh = h + kh - 1;
            if ((unsigned)hh >= 64u) continue;
#pragma unroll
            for (int kw = 0; kw < 3; ++kw) {
                int wi = w + kw - 1;
                if ((unsigned)wi >= 64u) continue;
                acc += wk[kh * 3 + kw] * tile[hh * 65 + wi];
            }
        }
        ap[j] = (__bf16)gelu_f(acc * iv1 + bt1);
    }
}

// ---------------------------------------------------------------------------
// K5: per-batch GEMM  out[o,n] = sum_c pwb[o,c]*act[c,n]  via
// v_wmma_f32_16x16x32_bf16, fused BN2 epilogue.
// Block tile 128x128, 8 waves (2M x 4N), wave tile 64x32 = 4x2 accumulators.
// A panel (128x32 bf16, pitch 40) is DMA'd by the Tensor Data Mover into a
// double buffer: D# with tile 32x128, data_size=2B, pad every 16 DWORDs by
// 4 DWORDs -> pitch-40 rows written directly by TDM; wave 0 issues the next
// panel after the barrier and synchronizes with s_wait_tensorcnt.
// B panel is transposed [n][k] via VALU so fragment loads are contiguous
// 16B-aligned ds_loads per the ISA 16-bit A(16x32)/B(32x16) lane layouts.
// ---------------------------------------------------------------------------
__global__ void __launch_bounds__(256)
k5_gemm(const __bf16* __restrict__ act, const __bf16* __restrict__ pwb,
        const float* bn2g, const float* bn2b, const float* bn2m, const float* bn2v,
        float* __restrict__ out) {
    const int nBase = blockIdx.x * 128;
    const int mBase = blockIdx.y * 128;
    const int b     = blockIdx.z;
    const int t = threadIdx.x;
    const unsigned lane = t & 31;
    const int wid = t >> 5, wm = wid >> 2, wn = wid & 3;
    __shared__ __align__(16) __bf16 ldsA[2][128 * 40];   // 2 x 10240 B
    __shared__ __align__(16) __bf16 ldsB[128 * 40];
    __shared__ float inv2[128], beta2[128];
    if (t < 128) {
        int o = mBase + t;
        float iv = bn2g[o] * rsqrtf(bn2v[o] + EPSF);
        inv2[t] = iv; beta2[t] = bn2b[o] - bn2m[o] * iv;
    }
    const __bf16* actB = act + (size_t)b * 1048576;
    v8f acc[4][2];
#pragma unroll
    for (int mt = 0; mt < 4; ++mt)
#pragma unroll
        for (int nt = 0; nt < 2; ++nt)
            acc[mt][nt] = (v8f){0.f, 0.f, 0.f, 0.f, 0.f, 0.f, 0.f, 0.f};

    const int bcd = t >> 3, bn0 = (t & 7) * 16;  // B staging: 16 elems/thread

#if HAS_TDM
    // D# group1 (invariant): data_size=2B, pad_enable, pad 4 DW every 16 DW,
    // tensor_dim0/1 = 2^20, tile_dim0 = 32 (k), tile_dim1 = 128 (m),
    // tensor_dim0_stride = 256 elements.
    const i32x8 g1 = { (int)((1u << 16) | (1u << 20) | (3u << 22) | (3u << 25)),
                       0,                         // abar=0, td0[15:0]=0
                       16,                        // td0[31:16]=16, td1[15:0]=0
                       (int)(16u | (32u << 16)),  // td1[31:16]=16, tile_dim0=32
                       128,                       // tile_dim1=128, tile_dim2=0
                       256,                       // dim0 stride (low 32)
                       0, 0 };
    const i32x4 gz4 = {0, 0, 0, 0};
    const i32x8 gz8 = {0, 0, 0, 0, 0, 0, 0, 0};
    const unsigned long long gaBase =
        (unsigned long long)(uintptr_t)(const void*)(pwb + (size_t)mBase * 256);
    const unsigned ldsBase = (unsigned)(uintptr_t)(void*)&ldsA[0][0];
    if (wid == 0) {                               // prologue: panel kk=0
        u32x4 g0 = { 1u, ldsBase,
                     (unsigned)(gaBase & 0xFFFFFFFFu),
                     (unsigned)((gaBase >> 32) & 0x1FFFFFFu) | (2u << 30) };
        __builtin_amdgcn_tensor_load_to_lds(g0, g1, gz4, gz4, gz8, 0);
    }
#else
    const int am = t >> 1, aks = (t & 1) * 16;   // A staging: 16 elems/thread
#endif

    for (int kk = 0; kk < 8; ++kk) {
        const int k = kk * 32;
        const __bf16* bsrc = actB + (size_t)(k + bcd) * 4096 + nBase + bn0;
        if (kk < 7) __builtin_prefetch(bsrc + 32 * 4096, 0, 1);
#pragma unroll
        for (int j = 0; j < 16; ++j) ldsB[(bn0 + j) * 40 + bcd] = bsrc[j];
#if HAS_TDM
        if (wid == 0) __builtin_amdgcn_s_wait_tensorcnt(0);   // A[kk] landed
#else
        const __bf16* asrc = pwb + (size_t)(mBase + am) * 256 + k + aks;
#pragma unroll
        for (int j = 0; j < 16; ++j) ldsA[kk & 1][am * 40 + aks + j] = asrc[j];
#endif
        __syncthreads();
#if HAS_TDM
        if (wid == 0 && kk < 7) {                 // overlap next A panel
            unsigned long long ga = gaBase + (unsigned long long)(k + 32) * 2u;
            u32x4 g0 = { 1u, ldsBase + (unsigned)(((kk + 1) & 1) * 10240),
                         (unsigned)(ga & 0xFFFFFFFFu),
                         (unsigned)((ga >> 32) & 0x1FFFFFFu) | (2u << 30) };
            __builtin_amdgcn_tensor_load_to_lds(g0, g1, gz4, gz4, gz8, 0);
        }
#endif
        const __bf16* Abuf = ldsA[kk & 1];
        // Fragment loads per the ISA 16-bit A(16x32)/B(32x16) lane layouts.
        const int k0 = (lane < 16) ? 0 : 8;    // A: K 0-7,16-23 | 8-15,24-31
        const int kb = (lane < 16) ? 0 : 8;    // B: K 0-7,16-23 | 8-15,24-31
        v16bf af[4], bfr[2];
#pragma unroll
        for (int mt = 0; mt < 4; ++mt) {
            int ml = wm * 64 + mt * 16 + (int)(lane & 15);
            v8bf lo = *(const v8bf*)(Abuf + ml * 40 + k0);
            v8bf hi = *(const v8bf*)(Abuf + ml * 40 + k0 + 16);
            af[mt] = __builtin_shufflevector(lo, hi, 0,1,2,3,4,5,6,7,8,9,10,11,12,13,14,15);
        }
#pragma unroll
        for (int nt = 0; nt < 2; ++nt) {
            int nl = wn * 32 + nt * 16 + (int)(lane & 15);
            v8bf lo = *(const v8bf*)(ldsB + nl * 40 + kb);
            v8bf hi = *(const v8bf*)(ldsB + nl * 40 + kb + 16);
            bfr[nt] = __builtin_shufflevector(lo, hi, 0,1,2,3,4,5,6,7,8,9,10,11,12,13,14,15);
        }
#pragma unroll
        for (int mt = 0; mt < 4; ++mt)
#pragma unroll
            for (int nt = 0; nt < 2; ++nt)
                acc[mt][nt] = __builtin_amdgcn_wmma_f32_16x16x32_bf16(
                    false, af[mt], false, bfr[nt], (short)0, acc[mt][nt],
                    false, false);
        __syncthreads();
    }

    // Epilogue: BN2 + store. C/D layout: lane<16 -> M = j, lane>=16 -> M = j+8.
    const int mofs = (lane < 16) ? 0 : 8;
    const int nl15 = (int)(lane & 15);
#pragma unroll
    for (int mt = 0; mt < 4; ++mt) {
#pragma unroll
        for (int nt = 0; nt < 2; ++nt) {
            int nl = wn * 32 + nt * 16 + nl15;
#pragma unroll
            for (int j = 0; j < 8; ++j) {
                int ml = wm * 64 + mt * 16 + j + mofs;
                float v = acc[mt][nt][j] * inv2[ml] + beta2[ml];
                out[((size_t)b * 256 + mBase + ml) * 4096 + nBase + nl] = v;
            }
        }
    }
}

// ---------------------------------------------------------------------------
extern "C" void kernel_launch(void* const* d_in, const int* in_sizes, int n_in,
                              void* d_out, int out_size, void* d_ws, size_t ws_size,
                              hipStream_t stream) {
    (void)in_sizes; (void)n_in; (void)out_size; (void)ws_size;
    const float* x     = (const float*)d_in[0];
    const float* adww  = (const float*)d_in[1];
    const float* adwb  = (const float*)d_in[2];
    const float* c4w   = (const float*)d_in[3];
    const float* c4b   = (const float*)d_in[4];
    const float* w41   = (const float*)d_in[5];
    const float* b41   = (const float*)d_in[6];
    const float* kgw1  = (const float*)d_in[7];
    const float* kgb1  = (const float*)d_in[8];
    const float* kgw2  = (const float*)d_in[9];
    const float* kgb2  = (const float*)d_in[10];
    const float* dxw   = (const float*)d_in[11];
    const float* dyw   = (const float*)d_in[12];
    const float* prw1  = (const float*)d_in[13];
    const float* prb1  = (const float*)d_in[14];
    const float* prw2  = (const float*)d_in[15];
    const float* prb2  = (const float*)d_in[16];
    const float* gcw1  = (const float*)d_in[17];
    const float* gcb1  = (const float*)d_in[18];
    const float* gcw2  = (const float*)d_in[19];
    const float* gcb2  = (const float*)d_in[20];
    const float* dww   = (const float*)d_in[21];
    const float* bn1g  = (const float*)d_in[22];
    const float* bn1b  = (const float*)d_in[23];
    const float* bn1m  = (const float*)d_in[24];
    const float* bn1v  = (const float*)d_in[25];
    const float* pww   = (const float*)d_in[26];
    const float* bn2g  = (const float*)d_in[27];
    const float* bn2b  = (const float*)d_in[28];
    const float* bn2m  = (const float*)d_in[29];
    const float* bn2v  = (const float*)d_in[30];

    char* wsb = (char*)d_ws;
    float*  gap   = (float*)(wsb + 0);                 //  32 KB
    float*  cha   = (float*)(wsb + 32768);             //  32 KB
    float*  xpool = (float*)(wsb + 65536);             //   2 MB
    float*  ypool = (float*)(wsb + 2162688);           //   2 MB
    float*  xline = (float*)(wsb + 4259840);           //   8 KB
    float*  yline = (float*)(wsb + 4268032);           //   8 KB
    float*  satt  = (float*)(wsb + 4276224);           // 512 KB
    __bf16* actb  = (__bf16*)(wsb + 4800512);          //  64 MB bf16 activations
    __bf16* pwb   = (__bf16*)(wsb + 71909376);         // 128 KB bf16 pw weight

    k0_cvt<<<256, 256, 0, stream>>>(pww, pwb);
    k1_reduce<<<32 * 256, 256, 0, stream>>>(x, xpool, ypool, gap);
    k2_small<<<32, 256, 0, stream>>>(xpool, ypool, gap, adww, adwb, c4w, c4b,
                                     w41, b41, kgw1, kgb1, kgw2, kgb2,
                                     dxw, dyw, gcw1, gcb1, gcw2, gcb2,
                                     xline, yline, cha);
    k3_spatial<<<32, 256, 0, stream>>>(xline, yline, prw1, prb1, prw2, prb2, satt);
    k4_fuse_dw<<<32 * 256, 256, 0, stream>>>(x, satt, cha, dww,
                                             bn1g, bn1b, bn1m, bn1v, actb);
    dim3 g5(32, 2, 32);   // N-tiles x M-tiles x batch
    k5_gemm<<<g5, 256, 0, stream>>>(actb, pwb, bn2g, bn2b, bn2m, bn2v,
                                    (float*)d_out);
}
